// Gemma3nAudioAttention_88441966559357
// MI455X (gfx1250) — matrix operand order repaired and checked by reading, not compiled, not run on hardware
//
#include <hip/hip_runtime.h>
#include <hip/hip_bf16.h>
#include <math.h>

typedef __attribute__((ext_vector_type(16))) _Float16 v16h;
typedef __attribute__((ext_vector_type(8)))  _Float16 h8;
typedef __attribute__((ext_vector_type(8)))  float    v8f;
typedef __attribute__((ext_vector_type(4)))  float    f4;

#define HID   1536
#define NHEAD 8
#define HD    192            // head dim
#define WBLK  12             // query chunk W
#define CTX   24             // key context C
#define FSPAN 13             // relative span F
#define MAXPAST 12
#define BB    4
#define TT    4096
#define UB    ((TT + WBLK - 1) / WBLK)   // 342
#define CAPV  50.0f
#define FMINF32 (-3.4028234663852886e38f)

// ---------------------------------------------------------------------------
// Kernel 1: fused QKV projection GEMM, f16 WMMA, f32 accumulate.
// C[m,n] = hidden[m,:] @ W[:,n];  z selects wq/wk/wv; q gets per-dim scale.
// Workgroup tile 128x128, 8 waves (wave32), wave tile 32x64 = 2x4 WMMA frags.
// ---------------------------------------------------------------------------
__global__ __launch_bounds__(256)
void qkv_gemm_f16(const float* __restrict__ hid,
                  const float* __restrict__ wq,
                  const float* __restrict__ wk,
                  const float* __restrict__ wv,
                  const float* __restrict__ pds,
                  _Float16* __restrict__ qkv)
{
    const int mt = blockIdx.x * 128;
    const int nt = blockIdx.y * 128;
    const int z  = blockIdx.z;
    const float* Wp = (z == 0) ? wq : (z == 1) ? wk : wv;
    _Float16* out = qkv + (size_t)z * (size_t)(BB * TT) * HID;

    __shared__ _Float16 As[128][40];   // 40 halves/row -> 80B stride (16B aligned)
    __shared__ _Float16 Bs[128][40];   // stored transposed: Bs[n][k]

    const int tid  = threadIdx.x;
    const int lane = tid & 31;
    const int wave = tid >> 5;
    const int wm   = (wave & 3)  * 32;   // wave row base within tile
    const int wn   = (wave >> 2) * 64;   // wave col base within tile

    const int rlo  = lane & 15;
    const bool hiL = lane >= 16;

    v8f acc[2][4];
#pragma unroll
    for (int i = 0; i < 2; i++)
#pragma unroll
        for (int j = 0; j < 4; j++)
#pragma unroll
            for (int r = 0; r < 8; r++) acc[i][j][r] = 0.0f;

    // staging assignments: A: thread -> (row = tid/2, 16 cols); B: (k = tid/8, 16 n's)
    const int arow = tid >> 1;
    const int acol = (tid & 1) * 16;
    const int bk   = tid >> 3;
    const int bn0  = (tid & 7) * 16;

    for (int kc = 0; kc < HID; kc += 32) {
        __syncthreads();
        {   // stage A (convert f32 -> f16), coalesced
            const float* src = hid + (size_t)(mt + arow) * HID + kc + acol;
#pragma unroll
            for (int v = 0; v < 4; v++) {
                f4 x = *(const f4*)(src + v * 4);
                As[arow][acol + v*4 + 0] = (_Float16)x.x;
                As[arow][acol + v*4 + 1] = (_Float16)x.y;
                As[arow][acol + v*4 + 2] = (_Float16)x.z;
                As[arow][acol + v*4 + 3] = (_Float16)x.w;
            }
        }
        {   // stage B transposed (Bs[n][k] = W[kc+k][nt+n]), coalesced global reads
            const float* src = Wp + (size_t)(kc + bk) * HID + nt + bn0;
#pragma unroll
            for (int v = 0; v < 4; v++) {
                f4 x = *(const f4*)(src + v * 4);
                Bs[bn0 + v*4 + 0][bk] = (_Float16)x.x;
                Bs[bn0 + v*4 + 1][bk] = (_Float16)x.y;
                Bs[bn0 + v*4 + 2][bk] = (_Float16)x.z;
                Bs[bn0 + v*4 + 3][bk] = (_Float16)x.w;
            }
        }
        __syncthreads();

        // A fragments (16x32): lane<16 -> row rlo, K{0..7,16..23}; lane>=16 -> K{8..15,24..31}
        v16h afr[2];
#pragma unroll
        for (int mf = 0; mf < 2; mf++) {
            int row = wm + mf * 16 + rlo;
            h8 lo = *(const h8*)&As[row][hiL ? 8  : 0];
            h8 hi = *(const h8*)&As[row][hiL ? 24 : 16];
#pragma unroll
            for (int e = 0; e < 8; e++) { afr[mf][e] = lo[e]; afr[mf][e + 8] = hi[e]; }
        }
        // B fragments (32x16): lane<16 -> column rlo, K{0..7,16..23}; lane>=16 -> K{8..15,24..31}
#pragma unroll
        for (int nf = 0; nf < 4; nf++) {
            int col = wn + nf * 16 + rlo;
            h8 lo = *(const h8*)&Bs[col][hiL ? 8  : 0];
            h8 hi = *(const h8*)&Bs[col][hiL ? 24 : 16];
            v16h bfr;
#pragma unroll
            for (int e = 0; e < 8; e++) { bfr[e] = lo[e]; bfr[e + 8] = hi[e]; }
            acc[0][nf] = __builtin_amdgcn_wmma_f32_16x16x32_f16(
                false, afr[0], false, bfr, (short)0, acc[0][nf], false, false);
            acc[1][nf] = __builtin_amdgcn_wmma_f32_16x16x32_f16(
                false, afr[1], false, bfr, (short)0, acc[1][nf], false, false);
        }
    }

    // epilogue: q-scale fused for z==0; store f16
#pragma unroll
    for (int nf = 0; nf < 4; nf++) {
        int n = nt + wn + nf * 16 + rlo;
        float scale = 1.0f;
        if (z == 0) {
            float x  = pds[n % HD];
            float sp = (x > 20.0f) ? x : log1pf(__expf(x));   // softplus
            scale = 0.10411692f * sp;                          // (192^-0.5)/ln2
        }
#pragma unroll
        for (int mf = 0; mf < 2; mf++) {
#pragma unroll
            for (int r = 0; r < 8; r++) {
                int m = mt + wm + mf * 16 + r + (hiL ? 8 : 0);
                out[(size_t)m * HID + n] = (_Float16)(acc[mf][nf][r] * scale);
            }
        }
    }
}

// ---------------------------------------------------------------------------
// Kernel 2: positional embedding projection: semb[f, n] = ts(f,:) @ w_pos[:, n]
// Tiny (13x1536x1536); coalesced across n.
// ---------------------------------------------------------------------------
__global__ void pos_emb_proj(const float* __restrict__ wpos,
                             _Float16* __restrict__ semb)
{
    int idx = blockIdx.x * blockDim.x + threadIdx.x;
    if (idx >= FSPAN * HID) return;
    int f = idx / HID, n = idx % HID;
    float pos = (float)(MAXPAST - f);
    const float log_inc = 9.210340371976184f / (float)(HID / 2 - 1); // ln(1e4)/767
    float a = 0.0f;
    for (int i = 0; i < HID; i++) {
        float ts;
        if (i < HID / 2) ts = __sinf(pos * __expf(-(float)i * log_inc));
        else             ts = __cosf(pos * __expf(-(float)(i - HID / 2) * log_inc));
        a += ts * wpos[(size_t)i * HID + n];
    }
    semb[(size_t)f * HID + n] = (_Float16)a;
}

// ---------------------------------------------------------------------------
// Kernel 3: blocked local attention, one wave32 per (u, h, b).
//  - WMMA content logits (2 N-tiles) + position logits (1 tile) over K=192
//  - scalar f32: XL relative shift, tanh cap, mask, softmax over C=24
//  - WMMA context: probs[16x32] @ v[32x192] (12 N-tiles)
// ---------------------------------------------------------------------------
__global__ __launch_bounds__(32)
void local_attn(const _Float16* __restrict__ qkv,
                const _Float16* __restrict__ semb,
                const unsigned char* __restrict__ mask,
                float* __restrict__ out)
{
    const int u = blockIdx.x, h = blockIdx.y, b = blockIdx.z;
    const int lane = threadIdx.x;
    const int rlo  = lane & 15;
    const bool hiL = lane >= 16;

    __shared__ _Float16 qs[16][200];   // q rows (w), zero-padded to 16
    __shared__ _Float16 ks[32][200];   // k rows (c), zero-padded to 32
    __shared__ _Float16 ss[16][200];   // sin-emb rows (f), zero-padded
    __shared__ _Float16 vT[HD][40];    // v transposed: vT[d][c]
    __shared__ float    acL[12][32];   // content logits
    __shared__ float    bdL[12][16];   // position logits (unshifted)
    __shared__ _Float16 ps[16][40];    // probs, A-operand layout friendly

    const _Float16* qg = qkv;
    const _Float16* kg = qkv + 1 * (size_t)(BB * TT) * HID;
    const _Float16* vg = qkv + 2 * (size_t)(BB * TT) * HID;

    for (int i = lane; i < 16 * 40; i += 32) ((_Float16*)ps)[i] = (_Float16)0.0f;

    // stage q (12 real rows), 8-half chunks
    for (int i = lane; i < 16 * 24; i += 32) {
        int w = i / 24, c8 = (i % 24) * 8;
        int t = u * WBLK + w;
        h8 val = {};
        if (w < WBLK && t < TT)
            val = *(const h8*)(qg + ((size_t)(b * TT + t) * HID) + h * HD + c8);
        *(h8*)&qs[w][c8] = val;
    }
    // stage k (24 real rows)
    for (int i = lane; i < 32 * 24; i += 32) {
        int c = i / 24, c8 = (i % 24) * 8;
        int t = u * WBLK + c - MAXPAST;
        h8 val = {};
        if (c < CTX && t >= 0 && t < TT)
            val = *(const h8*)(kg + ((size_t)(b * TT + t) * HID) + h * HD + c8);
        *(h8*)&ks[c][c8] = val;
    }
    // stage sin-emb (13 real rows)
    for (int i = lane; i < 16 * 24; i += 32) {
        int f = i / 24, c8 = (i % 24) * 8;
        h8 val = {};
        if (f < FSPAN)
            val = *(const h8*)(semb + (size_t)f * HID + h * HD + c8);
        *(h8*)&ss[f][c8] = val;
    }
    // stage v transposed
    for (int i = lane; i < 32 * 24; i += 32) {
        int c = i / 24, d0 = (i % 24) * 8;
        int t = u * WBLK + c - MAXPAST;
        h8 val = {};
        if (c < CTX && t >= 0 && t < TT)
            val = *(const h8*)(vg + ((size_t)(b * TT + t) * HID) + h * HD + d0);
#pragma unroll
        for (int j = 0; j < 8; j++) vT[d0 + j][c] = val[j];
    }
    __syncthreads();

    // ---- logits: term_ac (c 0..15 and 16..31) + term_bd, K loop over D=192 ----
    v8f acc0, acc1, accS;
#pragma unroll
    for (int r = 0; r < 8; r++) { acc0[r] = 0.f; acc1[r] = 0.f; accS[r] = 0.f; }

#pragma unroll
    for (int kc = 0; kc < HD; kc += 32) {
        v16h a;
        {
            h8 lo = *(const h8*)&qs[rlo][kc + (hiL ? 8  : 0)];
            h8 hi = *(const h8*)&qs[rlo][kc + (hiL ? 24 : 16)];
#pragma unroll
            for (int e = 0; e < 8; e++) { a[e] = lo[e]; a[e + 8] = hi[e]; }
        }
        v16h b0, b1, bs;
        {
            h8 lo = *(const h8*)&ks[rlo][kc + (hiL ? 8  : 0)];
            h8 hi = *(const h8*)&ks[rlo][kc + (hiL ? 24 : 16)];
#pragma unroll
            for (int e = 0; e < 8; e++) { b0[e] = lo[e]; b0[e + 8] = hi[e]; }
        }
        {
            h8 lo = *(const h8*)&ks[rlo + 16][kc + (hiL ? 8  : 0)];
            h8 hi = *(const h8*)&ks[rlo + 16][kc + (hiL ? 24 : 16)];
#pragma unroll
            for (int e = 0; e < 8; e++) { b1[e] = lo[e]; b1[e + 8] = hi[e]; }
        }
        {
            h8 lo = *(const h8*)&ss[rlo][kc + (hiL ? 8  : 0)];
            h8 hi = *(const h8*)&ss[rlo][kc + (hiL ? 24 : 16)];
#pragma unroll
            for (int e = 0; e < 8; e++) { bs[e] = lo[e]; bs[e + 8] = hi[e]; }
        }
        acc0 = __builtin_amdgcn_wmma_f32_16x16x32_f16(false, a, false, b0, (short)0, acc0, false, false);
        acc1 = __builtin_amdgcn_wmma_f32_16x16x32_f16(false, a, false, b1, (short)0, acc1, false, false);
        accS = __builtin_amdgcn_wmma_f32_16x16x32_f16(false, a, false, bs, (short)0, accS, false, false);
    }

    // dump logits (D layout: VGPR r -> row r / r+8, lane -> column)
#pragma unroll
    for (int r = 0; r < 8; r++) {
        int m = r + (hiL ? 8 : 0);
        if (m < WBLK) {
            acL[m][rlo]      = acc0[r];
            acL[m][rlo + 16] = acc1[r];
            bdL[m][rlo]      = accS[r];
        }
    }
    __syncthreads();

    // ---- relative shift + tanh cap + mask + softmax (f32, one row per lane) ----
    if (lane < WBLK) {
        int w = lane;
        float l[CTX];
        float mx = FMINF32;
#pragma unroll
        for (int c = 0; c < CTX; c++) {
            int p    = w * CTX + c;
            int wsrc = p / (CTX + 1);
            int fidx = p % (CTX + 1);
            float bd = (fidx < FSPAN) ? bdL[wsrc][fidx] : 0.0f;
            float raw = acL[w][c] + bd;
            float lg  = tanhf(raw * (1.0f / CAPV)) * CAPV;
            int t = u * WBLK + c - MAXPAST;
            bool valid = (t >= 0) && (t < TT) && (mask[b * TT + t] == 0)
                       && (w <= c) && (c <= w + MAXPAST);
            l[c] = valid ? lg : FMINF32;
            mx = fmaxf(mx, l[c]);
        }
        float s = 0.0f;
#pragma unroll
        for (int c = 0; c < CTX; c++) { l[c] = __expf(l[c] - mx); s += l[c]; }
        float inv = 1.0f / s;
#pragma unroll
        for (int c = 0; c < CTX; c++) ps[w][c] = (_Float16)(l[c] * inv);
    }
    __syncthreads();

    // ---- context: probs[16x32] @ v[32x192] via 12 WMMA N-tiles ----
    v16h ap;
    {
        h8 lo = *(const h8*)&ps[rlo][hiL ? 8  : 0];
        h8 hi = *(const h8*)&ps[rlo][hiL ? 24 : 16];
#pragma unroll
        for (int e = 0; e < 8; e++) { ap[e] = lo[e]; ap[e + 8] = hi[e]; }
    }
#pragma unroll
    for (int ntile = 0; ntile < HD / 16; ntile++) {
        int d = ntile * 16 + rlo;
        h8 lo = *(const h8*)&vT[d][hiL ? 8  : 0];
        h8 hi = *(const h8*)&vT[d][hiL ? 24 : 16];
        v16h bv;
#pragma unroll
        for (int e = 0; e < 8; e++) { bv[e] = lo[e]; bv[e + 8] = hi[e]; }
        v8f o;
#pragma unroll
        for (int r = 0; r < 8; r++) o[r] = 0.0f;
        o = __builtin_amdgcn_wmma_f32_16x16x32_f16(false, ap, false, bv, (short)0, o, false, false);
#pragma unroll
        for (int r = 0; r < 8; r++) {
            int m = r + (hiL ? 8 : 0);
            int t = u * WBLK + m;
            if (m < WBLK && t < TT)
                out[(((size_t)(b * TT + t)) * NHEAD + h) * HD + ntile * 16 + rlo] = o[r];
        }
    }
}

// ---------------------------------------------------------------------------
extern "C" void kernel_launch(void* const* d_in, const int* in_sizes, int n_in,
                              void* d_out, int out_size, void* d_ws, size_t ws_size,
                              hipStream_t stream)
{
    const float*         hid  = (const float*)d_in[0];
    const unsigned char* mask = (const unsigned char*)d_in[1];
    const float*         wq   = (const float*)d_in[2];
    const float*         wk   = (const float*)d_in[3];
    const float*         wv   = (const float*)d_in[4];
    const float*         wpos = (const float*)d_in[5];
    const float*         pds  = (const float*)d_in[6];

    _Float16* qkv  = (_Float16*)d_ws;                              // 3*16384*1536 f16
    _Float16* semb = qkv + (size_t)3 * (BB * TT) * HID;            // 13*1536 f16
    float*    out  = (float*)d_out;

    dim3 g1((BB * TT) / 128, HID / 128, 3);     // 128 x 12 x 3
    qkv_gemm_f16<<<g1, 256, 0, stream>>>(hid, wq, wk, wv, pds, qkv);

    pos_emb_proj<<<(FSPAN * HID + 255) / 256, 256, 0, stream>>>(wpos, semb);

    local_attn<<<dim3(UB, NHEAD, BB), 32, 0, stream>>>(qkv, semb, mask, out);
}
